// sSplitAttention_68736656605497
// MI455X (gfx1250) — matrix operand order repaired and checked by reading, not compiled, not run on hardware
//
#include <hip/hip_runtime.h>
#include <stdint.h>

// ---- problem constants ----
#define B_     2
#define N_     4096
#define DIM_   1024
#define H_     16
#define D_     64
#define LAT_   1024
#define CTX_   3072
#define SEG_   12
#define CHUNK_ 256
#define SCALE_ 0.125f          // 64^-0.5
#define NEGF   (-3.402823466e38f)

typedef __bf16 bf16_t;
typedef __attribute__((ext_vector_type(16))) __bf16    v16bf;
typedef __attribute__((ext_vector_type(8)))  float     v8f;
typedef __attribute__((ext_vector_type(4)))  unsigned  uv4;

union Pack16 { v16bf bf; uv4 q[2]; };

__device__ __forceinline__ v8f wmma_bf16(v16bf a, v16bf b, v8f c) {
  return __builtin_amdgcn_wmma_f32_16x16x32_bf16(false, a, false, b, (short)0, c,
                                                 false, false);
}

// A fragment 16x32 bf16 from fp32 row-major source at (m0,k0), leading dim ld.
// lane<16: row=lane, K = k0+{0..7, 16..23}; lane>=16: row=lane-16, K = k0+{8..15, 24..31}
__device__ __forceinline__ v16bf load_a_f32(const float* __restrict__ A, int ld) {
  const int lane = threadIdx.x & 31;
  const float* p = A + (size_t)(lane & 15) * ld + ((lane >> 4) << 3);
  v16bf a;
#pragma unroll
  for (int i = 0; i < 8; ++i) a[i] = (bf16_t)p[i];
#pragma unroll
  for (int i = 0; i < 8; ++i) a[8 + i] = (bf16_t)p[16 + i];
  return a;
}

// A fragment 16x32 bf16 from bf16 row-major source (global or LDS), 16B-aligned.
__device__ __forceinline__ v16bf load_a_bf16(const bf16_t* __restrict__ A, int ld) {
  const int lane = threadIdx.x & 31;
  const bf16_t* p = A + (size_t)(lane & 15) * ld + ((lane >> 4) << 3);
  Pack16 r;
  r.q[0] = *(const uv4*)p;
  r.q[1] = *(const uv4*)(p + 16);
  return r.bf;
}

// B fragment 32x16 bf16; source stored column-major: Bc[n*ldc + k] at (k0,n0).
// lane holds column n=lane&15, K = (lane>>4)*8 + {0..7, 16..23} (matches A).
__device__ __forceinline__ v16bf load_b_bf16(const bf16_t* __restrict__ Bc, int ldc) {
  const int lane = threadIdx.x & 31;
  const bf16_t* p = Bc + (size_t)(lane & 15) * ldc + ((lane >> 4) << 3);
  Pack16 r;
  r.q[0] = *(const uv4*)p;
  r.q[1] = *(const uv4*)(p + 16);
  return r.bf;
}

// ---------------- kernel 0: W (fp32 row-major KxN) -> WT (bf16 col-major N x K) ----
__global__ __launch_bounds__(256) void k_transpose_w(const float* __restrict__ W,
                                                     bf16_t* __restrict__ WT) {
  const int id = blockIdx.x * 256 + threadIdx.x;   // 1024*1024 elements
  const int n = id >> 10, k = id & 1023;
  WT[(size_t)n * 1024 + k] = (bf16_t)W[(size_t)k * 1024 + n];
}

// ---------------- kernel 1: C = A(fp32) x B, wave computes a 16x64 strip ----------
__global__ __launch_bounds__(128) void k_gemm_f32a(const float* __restrict__ A,
                                                   const bf16_t* __restrict__ BT,
                                                   float* __restrict__ C,
                                                   int M, int Nn, int K) {
  const int lane = threadIdx.x & 31, wid = threadIdx.x >> 5;
  const int wg = blockIdx.x * 4 + wid;
  const int strips_n = Nn >> 6;
  const int tm = wg / strips_n, sn = wg % strips_n;
  const float* Arow = A + (size_t)tm * 16 * K;
  v8f acc[4] = {};
  for (int k0 = 0; k0 < K; k0 += 32) {
    __builtin_prefetch(Arow + k0 + 128, 0, 0);
    v16bf a = load_a_f32(Arow + k0, K);
#pragma unroll
    for (int t = 0; t < 4; ++t) {
      v16bf b = load_b_bf16(BT + (size_t)(sn * 64 + t * 16) * K + k0, K);
      acc[t] = wmma_bf16(a, b, acc[t]);
    }
  }
  const int hi = lane >> 4, col = lane & 15;
#pragma unroll
  for (int t = 0; t < 4; ++t)
#pragma unroll
    for (int v = 0; v < 8; ++v)
      C[(size_t)(tm * 16 + v + 8 * hi) * Nn + sn * 64 + t * 16 + col] = acc[t][v];
}

// ---------------- kernel 2: RoPE(q)*scale, RoPE+LN(kv) -> bf16 q / lkv / lkvT ----
__global__ __launch_bounds__(256) void k_rope_ln(const float* __restrict__ qraw,
                                                 const float* __restrict__ kvraw,
                                                 const float* __restrict__ lng,
                                                 const float* __restrict__ lnb,
                                                 bf16_t* __restrict__ qbf,
                                                 bf16_t* __restrict__ lkv,
                                                 bf16_t* __restrict__ lkvT) {
  const int lane = threadIdx.x & 31, wid = threadIdx.x >> 5;
  const int row = blockIdx.x * 8 + wid;     // (bh, n), 131072 total
  const int bh = row >> 12, n = row & 4095;
  const int b = bh >> 4, h = bh & 15;
  const size_t src = ((size_t)b * N_ + n) * DIM_ + h * 64;
  const float t0q = qraw[src + lane], t1q = qraw[src + lane + 32];
  const float t0k = kvraw[src + lane], t1k = kvraw[src + lane + 32];
  const float inv = powf(10000.f, -(float)(2 * lane) * (1.f / 64.f));
  const float fr = (float)n * inv;
  const float cs = cosf(fr), si = sinf(fr);
  // rope: d<32: t*c - t[d+32]*s ; d>=32: t*c + t[d-32]*s
  const float q0 = t0q * cs - t1q * si;
  const float q1 = t1q * cs + t0q * si;
  const float k0 = t0k * cs - t1k * si;
  const float k1 = t1k * cs + t0k * si;
  const size_t qo = ((size_t)bh * N_ + n) * D_;
  qbf[qo + lane]      = (bf16_t)(q0 * SCALE_);
  qbf[qo + lane + 32] = (bf16_t)(q1 * SCALE_);
  // layernorm over D=64 (2 elems per lane, wave32 reduce)
  float mu = k0 + k1;
#pragma unroll
  for (int off = 16; off >= 1; off >>= 1) mu += __shfl_xor(mu, off, 32);
  mu *= (1.f / 64.f);
  const float d0 = k0 - mu, d1 = k1 - mu;
  float var = d0 * d0 + d1 * d1;
#pragma unroll
  for (int off = 16; off >= 1; off >>= 1) var += __shfl_xor(var, off, 32);
  var *= (1.f / 64.f);
  const float rs = rsqrtf(var + 1e-5f);
  const float o0 = d0 * rs * lng[lane] + lnb[lane];
  const float o1 = d1 * rs * lng[lane + 32] + lnb[lane + 32];
  lkv[qo + lane]      = (bf16_t)o0;
  lkv[qo + lane + 32] = (bf16_t)o1;
  const size_t to = (size_t)bh * D_ * N_;
  lkvT[to + (size_t)lane * N_ + n]        = (bf16_t)o0;
  lkvT[to + (size_t)(lane + 32) * N_ + n] = (bf16_t)o1;
}

// ---------------- kernels 3/4: flash attention (ctx segments / latent causal) ----
__global__ __launch_bounds__(128) void k_attn(const bf16_t* __restrict__ qbf,
                                              const bf16_t* __restrict__ lkv,
                                              const bf16_t* __restrict__ lkvT,
                                              bf16_t* __restrict__ obf,
                                              int is_latent) {
  __shared__ __align__(16) bf16_t stage[4][16 * 32];   // per-wave P staging (1KB each)
  const int lane = threadIdx.x & 31, wid = threadIdx.x >> 5;
  const int hi = lane >> 4, col = lane & 15;

  const int strip = blockIdx.x * 4 + wid;
  int bh, q0, k_begin, k_end;
  if (is_latent) {
    bh = strip >> 6;
    q0 = CTX_ + (strip & 63) * 16;
    k_begin = 0;
    k_end = q0 + 16;              // causal bound: keys j <= qi
  } else {
    bh = strip / (SEG_ * 16);
    const int r = strip % (SEG_ * 16);
    const int seg = r >> 4;
    q0 = seg * CHUNK_ + (r & 15) * 16;
    k_begin = seg * CHUNK_;
    k_end = k_begin + CHUNK_;
  }

  const bf16_t* qrow = qbf + ((size_t)bh * N_ + q0) * D_;
  const v16bf aQ0 = load_a_bf16(qrow, D_);        // d 0..31
  const v16bf aQ1 = load_a_bf16(qrow + 32, D_);   // d 32..63

  const bf16_t* kvn = lkv + (size_t)bh * N_ * D_;    // [n][d] -> B cols for QK^T
  const bf16_t* kvt = lkvT + (size_t)bh * D_ * N_;   // [d][n] -> B cols for P.V

  v8f acc[4] = {};
  float m[8], l[8];
#pragma unroll
  for (int v = 0; v < 8; ++v) { m[v] = NEGF; l[v] = 0.f; }

  bf16_t* stw = &stage[wid][0];

  for (int j0 = k_begin; j0 < k_end; j0 += 32) {
    // ---- S = (Q sc) . K^T for 32 keys: two 16x16 fragments ----
    v8f s0 = {}, s1 = {};
    {
      v16bf b = load_b_bf16(kvn + (size_t)j0 * D_, D_);
      s0 = wmma_bf16(aQ0, b, s0);
      b = load_b_bf16(kvn + (size_t)j0 * D_ + 32, D_);
      s0 = wmma_bf16(aQ1, b, s0);
      b = load_b_bf16(kvn + (size_t)(j0 + 16) * D_, D_);
      s1 = wmma_bf16(aQ0, b, s1);
      b = load_b_bf16(kvn + (size_t)(j0 + 16) * D_ + 32, D_);
      s1 = wmma_bf16(aQ1, b, s1);
    }
    // ---- causal mask among latents (only tail tiles can mask) ----
    if (is_latent && (j0 + 31 > q0)) {
#pragma unroll
      for (int v = 0; v < 8; ++v) {
        const int qi = q0 + v + 8 * hi;
        if (j0 + col > qi)      s0[v] = NEGF;
        if (j0 + 16 + col > qi) s1[v] = NEGF;
      }
    }
    // ---- online softmax update (rows replicated across each half-wave) ----
    float tm[8];
#pragma unroll
    for (int v = 0; v < 8; ++v) tm[v] = fmaxf(s0[v], s1[v]);
#pragma unroll
    for (int off = 8; off >= 1; off >>= 1)
#pragma unroll
      for (int v = 0; v < 8; ++v) tm[v] = fmaxf(tm[v], __shfl_xor(tm[v], off, 32));

    float sc[8], mn[8];
#pragma unroll
    for (int v = 0; v < 8; ++v) {
      mn[v] = fmaxf(m[v], tm[v]);
      sc[v] = __expf(m[v] - mn[v]);
      m[v] = mn[v];
    }
#pragma unroll
    for (int t = 0; t < 4; ++t)
#pragma unroll
      for (int v = 0; v < 8; ++v) acc[t][v] *= sc[v];

    float rs[8];
#pragma unroll
    for (int v = 0; v < 8; ++v) {
      const float p0 = __expf(s0[v] - mn[v]);
      const float p1 = __expf(s1[v] - mn[v]);
      rs[v] = p0 + p1;
      // C layout -> row-major P in LDS (row = v+8*hi, cols = col / col+16)
      stw[(v + 8 * hi) * 32 + col]      = (bf16_t)p0;
      stw[(v + 8 * hi) * 32 + 16 + col] = (bf16_t)p1;
    }
#pragma unroll
    for (int off = 8; off >= 1; off >>= 1)
#pragma unroll
      for (int v = 0; v < 8; ++v) rs[v] += __shfl_xor(rs[v], off, 32);
#pragma unroll
    for (int v = 0; v < 8; ++v) l[v] = l[v] * sc[v] + rs[v];

    // wait for per-wave LDS stores before re-reading as A fragment
    asm volatile("s_wait_dscnt 0" ::: "memory");
    const v16bf aP = load_a_bf16(stw, 32);
    asm volatile("" ::: "memory");

    // ---- O += P . V (K=32 keys, four 16-wide d tiles) ----
#pragma unroll
    for (int t = 0; t < 4; ++t) {
      v16bf bV = load_b_bf16(kvt + (size_t)(t * 16) * N_ + j0, N_);
      acc[t] = wmma_bf16(aP, bV, acc[t]);
    }
  }

  // ---- epilogue: normalize and store merged-head bf16 [b*N+n][h*64+d] ----
  const int b = bh >> 4, h = bh & 15;
  float il[8];
#pragma unroll
  for (int v = 0; v < 8; ++v) il[v] = 1.f / l[v];
#pragma unroll
  for (int t = 0; t < 4; ++t)
#pragma unroll
    for (int v = 0; v < 8; ++v) {
      const int n = q0 + v + 8 * hi;
      obf[((size_t)b * N_ + n) * DIM_ + h * 64 + t * 16 + col] =
          (bf16_t)(acc[t][v] * il[v]);
    }
}

// ---------------- kernel 5: out = attn(bf16) x Wo + bo, fp32 output --------------
__global__ __launch_bounds__(128) void k_gemm_bf16a(const bf16_t* __restrict__ A,
                                                    const bf16_t* __restrict__ BT,
                                                    const float* __restrict__ bias,
                                                    float* __restrict__ C,
                                                    int M, int Nn, int K) {
  const int lane = threadIdx.x & 31, wid = threadIdx.x >> 5;
  const int wg = blockIdx.x * 4 + wid;
  const int strips_n = Nn >> 6;
  const int tm = wg / strips_n, sn = wg % strips_n;
  const bf16_t* Arow = A + (size_t)tm * 16 * K;
  v8f acc[4] = {};
  for (int k0 = 0; k0 < K; k0 += 32) {
    __builtin_prefetch(Arow + k0 + 256, 0, 0);
    v16bf a = load_a_bf16(Arow + k0, K);
#pragma unroll
    for (int t = 0; t < 4; ++t) {
      v16bf b = load_b_bf16(BT + (size_t)(sn * 64 + t * 16) * K + k0, K);
      acc[t] = wmma_bf16(a, b, acc[t]);
    }
  }
  const int hi = lane >> 4, col = lane & 15;
#pragma unroll
  for (int t = 0; t < 4; ++t)
#pragma unroll
    for (int v = 0; v < 8; ++v) {
      const int nc = sn * 64 + t * 16 + col;
      C[(size_t)(tm * 16 + v + 8 * hi) * Nn + nc] = acc[t][v] + bias[nc];
    }
}

// ---------------- host-side orchestration ----------------------------------------
extern "C" void kernel_launch(void* const* d_in, const int* in_sizes, int n_in,
                              void* d_out, int out_size, void* d_ws, size_t ws_size,
                              hipStream_t stream) {
  const float* x    = (const float*)d_in[0];
  const float* Wq   = (const float*)d_in[1];
  const float* Wkv  = (const float*)d_in[2];
  const float* Wo   = (const float*)d_in[3];
  const float* bo   = (const float*)d_in[4];
  const float* ln_g = (const float*)d_in[5];
  const float* ln_b = (const float*)d_in[6];
  float* out = (float*)d_out;

  char* ws = (char*)d_ws;
  float*  q_raw  = (float*)(ws + 0);            // 33.5 MB (recycled as attn_bf)
  float*  kv_raw = (float*)(ws + (33554432u));  // 33.5 MB
  bf16_t* q_bf   = (bf16_t*)(ws + (67108864u)); // 16.8 MB
  bf16_t* lkv    = (bf16_t*)(ws + (83886080u)); // 16.8 MB
  bf16_t* lkvT   = (bf16_t*)(ws + (100663296u));// 16.8 MB
  bf16_t* WqT    = (bf16_t*)(ws + (117440512u));// 2 MB
  bf16_t* WkvT   = (bf16_t*)(ws + (119537664u));// 2 MB
  bf16_t* WoT    = (bf16_t*)(ws + (121634816u));// 2 MB
  bf16_t* attn_bf = (bf16_t*)(ws + 0);          // reuses q_raw region

  const int MR = B_ * N_;  // 8192

  // weights -> bf16, transposed (col-major B operands)
  k_transpose_w<<<4096, 256, 0, stream>>>(Wq, WqT);
  k_transpose_w<<<4096, 256, 0, stream>>>(Wkv, WkvT);
  k_transpose_w<<<4096, 256, 0, stream>>>(Wo, WoT);

  // q = x @ Wq ; kv = x @ Wkv   (512 m-tiles * 16 n-strips = 8192 waves)
  k_gemm_f32a<<<2048, 128, 0, stream>>>(x, WqT, q_raw, MR, DIM_, DIM_);
  k_gemm_f32a<<<2048, 128, 0, stream>>>(x, WkvT, kv_raw, MR, DIM_, DIM_);

  // rope + scale + layernorm -> bf16 head-major buffers
  k_rope_ln<<<16384, 256, 0, stream>>>(q_raw, kv_raw, ln_g, ln_b, q_bf, lkv, lkvT);

  // context segments (6144 strips) and latent causal (2048 strips)
  k_attn<<<1536, 128, 0, stream>>>(q_bf, lkv, lkvT, attn_bf, 0);
  k_attn<<<512, 128, 0, stream>>>(q_bf, lkv, lkvT, attn_bf, 1);

  // out = attn @ Wo + bo
  k_gemm_bf16a<<<2048, 128, 0, stream>>>(attn_bf, WoT, bo, out, MR, DIM_, DIM_);
}
